// Encoder_17549236371616
// MI455X (gfx1250) — compile-verified, matrix operand order checked by reading, not run on hardware
//
#include <hip/hip_runtime.h>
#include <math.h>

typedef __bf16 bf16;
typedef __attribute__((ext_vector_type(16))) __bf16 v16bf;
typedef __attribute__((ext_vector_type(8)))  float  v8f;
typedef __attribute__((ext_vector_type(4)))  unsigned int u32x4;
typedef __attribute__((ext_vector_type(4)))  int i32x4;
typedef __attribute__((ext_vector_type(8)))  int i32x8;

#define EPI_NONE      0
#define EPI_BIAS_RELU 1
#define EPI_BIAS_RES  2

// ---------------------------------------------------------------------------
// Tensor Data Mover: DMA `nelem` contiguous f32 from global to LDS offset
// `ldsOff` (dynamic-LDS base = 0).  1-D descriptor: tensor_dim0 = tile_dim0 =
// nelem, data_size = 4B, stride = nelem, no gather/iterate/multicast.
// Issue from one wave, wait TENSORcnt==0, then workgroup barrier outside.
// ---------------------------------------------------------------------------
__device__ __forceinline__ void tdm_copy_f32_to_lds(const float* __restrict__ g,
                                                    unsigned ldsOff, int nelem) {
    unsigned long long ga = (unsigned long long)(uintptr_t)g;
    u32x4 g0;
    g0[0] = 1u;                                                // count=1, user mode
    g0[1] = ldsOff;                                            // lds_addr (bytes)
    g0[2] = (unsigned)ga;                                      // global_addr[31:0]
    g0[3] = ((unsigned)(ga >> 32) & 0x01FFFFFFu) | 0x80000000u; // addr[56:32] | type=2
    i32x8 g1;
    g1[0] = (int)(2u << 16);                                   // data_size = 4B
    g1[1] = (int)(((unsigned)nelem & 0xFFFFu) << 16);          // tensor_dim0[15:0]
    g1[2] = (int)((((unsigned)nelem >> 16) & 0xFFFFu) | (1u << 16)); // dim0 hi | tensor_dim1=1
    g1[3] = (int)(((unsigned)nelem & 0xFFFFu) << 16);          // tile_dim0
    g1[4] = 1;                                                 // tile_dim1 = 1
    g1[5] = nelem;                                             // tensor_dim0_stride lo32
    g1[6] = 0;
    g1[7] = 0;
    i32x4 z4 = {0, 0, 0, 0};
#if __clang_major__ >= 23
    i32x8 z8 = {0, 0, 0, 0, 0, 0, 0, 0};
    __builtin_amdgcn_tensor_load_to_lds(g0, g1, z4, z4, z8, 0);
#else
    __builtin_amdgcn_tensor_load_to_lds(g0, g1, z4, z4, 0);
#endif
    __builtin_amdgcn_s_wait_tensorcnt(0);
}

// ---------------------------------------------------------------------------
// A fragment for V_WMMA_F32_16X16X32_BF16 (wave32), from a global f32 row:
//   element e<8  -> k = c*32 + hb*8 + e
//   element e>=8 -> k = c*32 + 16 + hb*8 + (e-8)
// ---------------------------------------------------------------------------
__device__ __forceinline__ v16bf load_a_frag_f32(const float* __restrict__ row,
                                                 int c, int hb) {
    v16bf a;
    const float4* p0 = (const float4*)(row + c * 32 + hb * 8);
    const float4* p1 = (const float4*)(row + c * 32 + 16 + hb * 8);
    float4 q0 = p0[0], q1 = p0[1], q2 = p1[0], q3 = p1[1];
    a[0] = (bf16)q0.x; a[1] = (bf16)q0.y; a[2]  = (bf16)q0.z; a[3]  = (bf16)q0.w;
    a[4] = (bf16)q1.x; a[5] = (bf16)q1.y; a[6]  = (bf16)q1.z; a[7]  = (bf16)q1.w;
    a[8] = (bf16)q2.x; a[9] = (bf16)q2.y; a[10] = (bf16)q2.z; a[11] = (bf16)q2.w;
    a[12]= (bf16)q3.x; a[13]= (bf16)q3.y; a[14] = (bf16)q3.z; a[15] = (bf16)q3.w;
    return a;
}

// ---------------------------------------------------------------------------
// Pack f32 weights (row-major KD x NC, staged in LDS) into per-lane-contiguous
// bf16 B-fragments: fragment (n,c), lane (hb,r) owns 16 bf16 at a 32B-aligned
// address -> the fragment load is a single contiguous 32-byte ds read.
// ---------------------------------------------------------------------------
template <int KD, int NC>
__device__ __forceinline__ void pack_weights(const float* __restrict__ sWf,
                                             bf16* __restrict__ sB, int tid) {
    constexpr int KC = KD / 32;
    for (int i = tid; i < KD * NC; i += 128) {
        int k = i / NC, col = i % NC;
        int n = col >> 4, r = col & 15;
        int c = k >> 5, kl = k & 31;
        int hb = (kl >> 3) & 1;
        int e  = (kl & 7) | ((kl >> 1) & 8);
        sB[((((n * KC + c) << 1) + hb) << 4 | r) * 16 + e] = (bf16)sWf[i];
    }
}

__device__ __forceinline__ v16bf load_b_packed(const bf16* __restrict__ sB,
                                               int frag, int r, int hb) {
    return *(const v16bf*)(sB + ((((frag << 1) + hb) << 4 | r) << 4));
}

// Butterfly sum over 16 lanes via ds_swizzle (group-of-32: imm=(xor<<10)|0x1F)
__device__ __forceinline__ float butterfly16_sum(float p) {
    p += __int_as_float(__builtin_amdgcn_ds_swizzle(__float_as_int(p), 0x041F));
    p += __int_as_float(__builtin_amdgcn_ds_swizzle(__float_as_int(p), 0x081F));
    p += __int_as_float(__builtin_amdgcn_ds_swizzle(__float_as_int(p), 0x101F));
    p += __int_as_float(__builtin_amdgcn_ds_swizzle(__float_as_int(p), 0x201F));
    return p;
}

// ---------------------------------------------------------------------------
// M x KD @ KD x NC GEMM.  Block = 128 threads = 4 waves; wave = 16 rows x NC.
// Dynamic LDS: [0, KD*NC*4) f32 staging (TDM target), then packed bf16 frags.
// ---------------------------------------------------------------------------
template <int KD, int NC, int EPI>
__global__ __launch_bounds__(128)
void gemm_wmma(const float* __restrict__ A, const float* __restrict__ W,
               const float* __restrict__ bias, const float* __restrict__ res,
               float* __restrict__ out, int M) {
    extern __shared__ char smem[];
    float* sWf = (float*)smem;
    bf16*  sB  = (bf16*)(smem + KD * NC * sizeof(float));
    const int tid = threadIdx.x;

    if (tid < 32) tdm_copy_f32_to_lds(W, 0u, KD * NC);
    __syncthreads();
    pack_weights<KD, NC>(sWf, sB, tid);
    __syncthreads();

    const int wave = tid >> 5, lane = tid & 31;
    const int r = lane & 15, hb = lane >> 4;
    const int row0 = (blockIdx.x * 4 + wave) * 16;
    if (row0 >= M) return;

    constexpr int NT = NC / 16, KC = KD / 32;

    int rowA = row0 + r; if (rowA > M - 1) rowA = M - 1;
    v16bf afrag[KC];
    #pragma unroll
    for (int c = 0; c < KC; ++c)
        afrag[c] = load_a_frag_f32(A + (long)rowA * KD, c, hb);

    v8f acc[NT] = {};
    #pragma unroll
    for (int n = 0; n < NT; ++n) {
        #pragma unroll
        for (int c = 0; c < KC; ++c) {
            v16bf bfrag = load_b_packed(sB, n * KC + c, r, hb);
            acc[n] = __builtin_amdgcn_wmma_f32_16x16x32_bf16(
                false, afrag[c], false, bfrag, (short)0, acc[n], false, false);
        }
    }

    #pragma unroll
    for (int n = 0; n < NT; ++n) {
        #pragma unroll
        for (int j = 0; j < 8; ++j) {
            int row = row0 + j + hb * 8;
            int col = n * 16 + r;
            if (row < M) {
                float v = acc[n][j];
                if (EPI != EPI_NONE) v += bias[col];
                if (EPI == EPI_BIAS_RELU) v = fmaxf(v, 0.0f);
                if (EPI == EPI_BIAS_RES)  v += res[(long)row * NC + col];
                out[(long)row * NC + col] = v;
            }
        }
    }
}

// ---------------------------------------------------------------------------
// Fused edge kernel: one wave = 16 edges.
//   Eh tile (16x64) = edge_attr tile @ WE via WMMA (head h == 16-col tile h)
//   s = sum_d K[src]*Q[dst]*Eh / 4 ; score = exp(clip(s,-5,5))
//   atomic scatter: wV[dst] += V[src]*score ; Z[dst,h] += score
// ---------------------------------------------------------------------------
__global__ __launch_bounds__(128)
void edge_attn_kernel(const float* __restrict__ edge_attr,
                      const int* __restrict__ srcIdx, const int* __restrict__ dstIdx,
                      const float* __restrict__ WE,
                      const float* __restrict__ Qh, const float* __restrict__ Kh,
                      const float* __restrict__ Vh,
                      float* __restrict__ wV, float* __restrict__ Zb, int Ecnt) {
    extern __shared__ char smem[];
    float* sWf = (float*)smem;
    bf16*  sB  = (bf16*)(smem + 64 * 64 * sizeof(float));
    const int tid = threadIdx.x;

    if (tid < 32) tdm_copy_f32_to_lds(WE, 0u, 64 * 64);
    __syncthreads();
    pack_weights<64, 64>(sWf, sB, tid);
    __syncthreads();

    const int wave = tid >> 5, lane = tid & 31;
    const int r = lane & 15, hb = lane >> 4;
    const long ebase = ((long)blockIdx.x * 4 + wave) * 16;
    if (ebase >= Ecnt) return;

    long erowA = ebase + r; if (erowA > Ecnt - 1) erowA = Ecnt - 1;
    v16bf afrag[2];
    afrag[0] = load_a_frag_f32(edge_attr + erowA * 64, 0, hb);
    afrag[1] = load_a_frag_f32(edge_attr + erowA * 64, 1, hb);

    int se[8], de[8];
    #pragma unroll
    for (int j = 0; j < 8; ++j) {
        long e = ebase + j + 8 * hb; if (e > Ecnt - 1) e = Ecnt - 1;
        se[j] = srcIdx[e];
        de[j] = dstIdx[e];
    }

    #pragma unroll
    for (int h = 0; h < 4; ++h) {
        v8f acc = {};
        #pragma unroll
        for (int c = 0; c < 2; ++c) {
            v16bf bfrag = load_b_packed(sB, h * 2 + c, r, hb);
            acc = __builtin_amdgcn_wmma_f32_16x16x32_bf16(
                false, afrag[c], false, bfrag, (short)0, acc, false, false);
        }
        const int col = h * 16 + r;
        #pragma unroll
        for (int j = 0; j < 8; ++j) {
            long e = ebase + j + 8 * hb;
            if (e >= Ecnt) continue;
            const int s = se[j], d = de[j];
            float p = acc[j] * Kh[(long)s * 64 + col] * Qh[(long)d * 64 + col];
            p = butterfly16_sum(p) * 0.25f;          // 1/sqrt(DH), DH=16
            p = fminf(fmaxf(p, -5.0f), 5.0f);
            p = __expf(p);
            atomicAdd(&wV[(long)d * 64 + col], Vh[(long)s * 64 + col] * p);
            if (r == 0) atomicAdd(&Zb[(long)d * 4 + h], p);
        }
    }
}

// h_attn = wV / (Z + 1e-6), in place
__global__ void attn_norm_kernel(float* __restrict__ wV, const float* __restrict__ Zb,
                                 int total) {
    int i = blockIdx.x * blockDim.x + threadIdx.x;
    if (i < total) {
        int row = i >> 6, head = (i >> 4) & 3;
        wV[i] = wV[i] / (Zb[row * 4 + head] + 1e-6f);
    }
}

// Row LayerNorm over 64 columns; one thread per row.
__global__ void ln_kernel(const float* __restrict__ in, const float* __restrict__ g,
                          const float* __restrict__ b, float* __restrict__ out, int M) {
    int row = blockIdx.x * blockDim.x + threadIdx.x;
    if (row >= M) return;
    const float4* p = (const float4*)(in + (long)row * 64);
    float4 v[16];
    float mu = 0.0f;
    #pragma unroll
    for (int i = 0; i < 16; ++i) {
        v[i] = p[i];
        mu += v[i].x + v[i].y + v[i].z + v[i].w;
    }
    mu *= (1.0f / 64.0f);
    float var = 0.0f;
    #pragma unroll
    for (int i = 0; i < 16; ++i) {
        float dx = v[i].x - mu, dy = v[i].y - mu, dz = v[i].z - mu, dw = v[i].w - mu;
        var += dx * dx + dy * dy + dz * dz + dw * dw;
    }
    float inv = rsqrtf(var * (1.0f / 64.0f) + 1e-5f);
    const float4* gp = (const float4*)g;
    const float4* bp = (const float4*)b;
    float4* op = (float4*)(out + (long)row * 64);
    #pragma unroll
    for (int i = 0; i < 16; ++i) {
        float4 gg = gp[i], bb = bp[i], o;
        o.x = (v[i].x - mu) * inv * gg.x + bb.x;
        o.y = (v[i].y - mu) * inv * gg.y + bb.y;
        o.z = (v[i].z - mu) * inv * gg.z + bb.z;
        o.w = (v[i].w - mu) * inv * gg.w + bb.w;
        op[i] = o;
    }
}

// ---------------------------------------------------------------------------
extern "C" void kernel_launch(void* const* d_in, const int* in_sizes, int n_in,
                              void* d_out, int out_size, void* d_ws, size_t ws_size,
                              hipStream_t stream) {
    (void)n_in; (void)out_size; (void)ws_size;
    const float* x    = (const float*)d_in[0];
    const float* eatt = (const float*)d_in[1];
    const int*   eidx = (const int*)d_in[2];
    const float* WQ   = (const float*)d_in[3];
    const float* WK   = (const float*)d_in[4];
    const float* WE   = (const float*)d_in[5];
    const float* WV   = (const float*)d_in[6];
    const float* WO   = (const float*)d_in[7];
    const float* bO   = (const float*)d_in[8];
    const float* g1   = (const float*)d_in[9];
    const float* be1  = (const float*)d_in[10];
    const float* W1   = (const float*)d_in[11];
    const float* b1   = (const float*)d_in[12];
    const float* W2   = (const float*)d_in[13];
    const float* b2   = (const float*)d_in[14];
    const float* g2   = (const float*)d_in[15];
    const float* be2  = (const float*)d_in[16];

    const int N = in_sizes[0] / 64;
    const int E = in_sizes[2] / 2;
    const int* srcI = eidx;
    const int* dstI = eidx + E;

    float* ws   = (float*)d_ws;
    float* Qh   = ws;                     // N*64
    float* Kh   = Qh   + (long)N * 64;
    float* Vh   = Kh   + (long)N * 64;
    float* wV   = Vh   + (long)N * 64;
    float* Zb   = wV   + (long)N * 64;    // N*4
    float* tmp  = Zb   + (long)N * 4;     // N*64
    float* mid  = tmp  + (long)N * 64;    // N*128
    float* hbuf = mid  + (long)N * 128;   // N*64

    hipMemcpyAsync(hbuf, x, (size_t)N * 64 * sizeof(float),
                   hipMemcpyDeviceToDevice, stream);

    const int gN  = (N + 63) / 64;
    const int gE  = (E + 63) / 64;
    const int gEl = (N * 64 + 255) / 256;
    const int gLn = (N + 255) / 256;

    const size_t smem64  = 64 * 64 * 4 + 64 * 64 * 2;     // 24 KB
    const size_t smem128 = 64 * 128 * 4 + 64 * 128 * 2;   // 48 KB

    for (int l = 0; l < 2; ++l) {
        const float* wq = WQ + (long)l * 4096;
        const float* wk = WK + (long)l * 4096;
        const float* we = WE + (long)l * 4096;
        const float* wv = WV + (long)l * 4096;
        const float* wo = WO + (long)l * 4096;
        const float* w1 = W1 + (long)l * 64 * 128;
        const float* w2 = W2 + (long)l * 128 * 64;

        hipMemsetAsync(wV, 0, (size_t)N * 64 * sizeof(float), stream);
        hipMemsetAsync(Zb, 0, (size_t)N * 4 * sizeof(float), stream);

        gemm_wmma<64, 64, EPI_NONE><<<gN, 128, smem64, stream>>>(hbuf, wq, nullptr, nullptr, Qh, N);
        gemm_wmma<64, 64, EPI_NONE><<<gN, 128, smem64, stream>>>(hbuf, wk, nullptr, nullptr, Kh, N);
        gemm_wmma<64, 64, EPI_NONE><<<gN, 128, smem64, stream>>>(hbuf, wv, nullptr, nullptr, Vh, N);

        edge_attn_kernel<<<gE, 128, smem64, stream>>>(eatt, srcI, dstI, we, Qh, Kh, Vh, wV, Zb, E);

        attn_norm_kernel<<<gEl, 256, 0, stream>>>(wV, Zb, N * 64);

        gemm_wmma<64, 64, EPI_BIAS_RES><<<gN, 128, smem64, stream>>>(
            wV, wo, bO + l * 64, hbuf, tmp, N);
        ln_kernel<<<gLn, 256, 0, stream>>>(tmp, g1 + l * 64, be1 + l * 64, Qh, N); // Qh := h1

        gemm_wmma<64, 128, EPI_BIAS_RELU><<<gN, 128, smem128, stream>>>(
            Qh, w1, b1 + l * 128, nullptr, mid, N);
        gemm_wmma<128, 64, EPI_BIAS_RES><<<gN, 128, smem128, stream>>>(
            mid, w2, b2 + l * 64, Qh, tmp, N);

        float* outp = (l == 1) ? (float*)d_out : hbuf;
        ln_kernel<<<gLn, 256, 0, stream>>>(tmp, g2 + l * 64, be2 + l * 64, outp, N);
    }
}
